// ParallelLSTM_21955872817556
// MI455X (gfx1250) — compile-verified, matrix operand order checked by reading, not run on hardware
//
#include <hip/hip_runtime.h>
#include <hip/hip_bf16.h>

typedef __attribute__((ext_vector_type(16))) __bf16 v16bf;
typedef __attribute__((ext_vector_type(8)))  float  v8f;
typedef unsigned short ushort_t;

constexpr int kL = 4, kB = 16, kT = 1024, kD = 512, k4D = 2048;
constexpr int kKcat   = 2 * kD;          // fused K: [x_t ; h_{t-1}] = 1024
constexpr int kNWG    = 16;              // workgroups per layer
constexpr int kCols   = kD / kNWG;       // 32 h-columns owned per WG
constexpr int kThreads = 128;            // 4 waves, one per gate (i,f,g,o)

// workspace layout (bytes)
constexpr size_t XBF_OFF  = 0;
constexpr size_t XBF_SZ   = (size_t)kB * kT * kD * 2;          // x in bf16, 16 MiB
constexpr size_t WCAT_OFF = XBF_OFF + XBF_SZ;
constexpr size_t WCAT_SZ  = (size_t)kL * k4D * kKcat * 2;      // [W;U]^T bf16, 16 MiB
constexpr size_t HBUF_OFF = WCAT_OFF + WCAT_SZ;
constexpr size_t HBUF_SZ  = (size_t)2 * kL * kB * kD * 2;      // ping-pong h, bf16
constexpr size_t BAR_OFF  = HBUF_OFF + HBUF_SZ;                // kL u32 counters

// d_out layout (f32): out_embd [B,T,D] ++ hT [L,B,D] ++ cT [L,B,D]
constexpr size_t HT_OFF = (size_t)kB * kT * kD;
constexpr size_t CT_OFF = HT_OFF + (size_t)kL * kB * kD;

// dynamic LDS: weight slice + gate-exchange + cell state
constexpr int SMEM_BYTES = 4 * kCols * kKcat * 2      // 262144 B bf16 weights
                         + 4 * kB * kCols * 4         //   8192 B z exchange
                         + kB * kCols * 4;            //   2048 B cell state

__device__ inline ushort_t f2bf(float f) {   // round-to-nearest-even f32 -> bf16
    unsigned u = __float_as_uint(f);
    u += 0x7FFFu + ((u >> 16) & 1u);
    return (ushort_t)(u >> 16);
}

// CDNA5 hardware tanh (v_tanh_f32); sigmoid via tanh identity -> 1 TRANS + 1 FMA
#if __has_builtin(__builtin_amdgcn_tanhf)
__device__ inline float fast_tanh(float x) { return __builtin_amdgcn_tanhf(x); }
#else
__device__ inline float fast_tanh(float x) { return tanhf(x); }
#endif
__device__ inline float fast_sigmoid(float x) {
    return __builtin_fmaf(0.5f, fast_tanh(0.5f * x), 0.5f);
}

__global__ void prep_zero(float* out, ushort_t* hbuf, unsigned* bar) {
    const size_t stride = (size_t)gridDim.x * blockDim.x;
    size_t i = (size_t)blockIdx.x * blockDim.x + threadIdx.x;
    for (size_t j = i; j < (size_t)kB * kT * kD; j += stride) out[j] = 0.f;
    for (size_t j = i; j < (size_t)2 * kL * kB * kD; j += stride) hbuf[j] = 0;
    if (i < (size_t)kL) bar[i] = 0u;   // reset barrier every call (graph replays)
}

__global__ void cvt_x(const float* __restrict__ x, ushort_t* __restrict__ xbf) {
    size_t i = (size_t)blockIdx.x * blockDim.x + threadIdx.x;
    if (i < (size_t)kB * kT * kD) xbf[i] = f2bf(x[i]);
}

// wcat[l][n][k] = (k < D ? kernel[l][k][n] : rec_kernel[l][k-D][n]) as bf16 (N-major)
__global__ void cvt_w(const float* __restrict__ kern, const float* __restrict__ rec,
                      ushort_t* __restrict__ wcat) {
    size_t i = (size_t)blockIdx.x * blockDim.x + threadIdx.x;
    if (i >= (size_t)kL * k4D * kKcat) return;
    const int k = (int)(i & (kKcat - 1));
    const int n = (int)((i >> 10) & (k4D - 1));
    const int l = (int)(i >> 21);
    float v = (k < kD) ? kern[((size_t)l * kD + k) * k4D + n]
                       : rec [((size_t)l * kD + (k - kD)) * k4D + n];
    wcat[i] = f2bf(v);
}

__global__ void __launch_bounds__(kThreads, 1)
lstm_scan(const ushort_t* __restrict__ xbf, const ushort_t* __restrict__ wcat,
          const float* __restrict__ bias, ushort_t* __restrict__ hbuf,
          unsigned* __restrict__ bar, float* __restrict__ out) {
    extern __shared__ ushort_t smem[];
    ushort_t* w_lds = smem;                                     // [4][kCols][kKcat]
    float*    z_lds = (float*)(smem + 4 * kCols * kKcat);       // [4][kB][kCols]
    float*    c_lds = z_lds + 4 * kB * kCols;                   // [kB][kCols]

    const int l  = blockIdx.x / kNWG;
    const int d0 = (blockIdx.x % kNWG) * kCols;
    const int tid  = threadIdx.x;
    const int w    = tid >> 5;       // wave index == gate index (i,f,g,o)
    const int lane = tid & 31;
    const int ln   = lane & 15;
    const int hi   = lane >> 4;

    // ---- stage this WG's 256 KiB weight slice into LDS (reused for all 1024 steps)
    {
        const int CH = kKcat / 8;    // 128 float4 chunks per weight row
        const float4* src = (const float4*)(wcat + (size_t)l * k4D * kKcat);
        float4* dst = (float4*)w_lds;
        for (int i = tid; i < 4 * kCols * CH; i += kThreads) {
            const int g   = i / (kCols * CH);
            const int rem = i - g * (kCols * CH);
            const int c   = rem / CH;
            const int k8  = rem - c * CH;
            const int n   = g * kD + d0 + c;
            dst[i] = src[(size_t)n * CH + k8];
        }
        for (int i = tid; i < kB * kCols; i += kThreads) c_lds[i] = 0.f;
    }
    __syncthreads();

    // bias broadcast per lane (same N for all 8 accumulator rows)
    const float b0 = bias[(size_t)l * k4D + w * kD + d0 + ln];
    const float b1 = bias[(size_t)l * k4D + w * kD + d0 + 16 + ln];

    const ushort_t* wrow0 = w_lds + (size_t)(w * kCols + ln) * kKcat;
    const ushort_t* wrow1 = wrow0 + (size_t)16 * kKcat;

    for (int t = 0; t < kT; ++t) {
        const ushort_t* hprev = hbuf + ((size_t)(t & 1) * kL + l) * kB * kD;
        ushort_t*       hnext = hbuf + ((size_t)((t + 1) & 1) * kL + l) * kB * kD;

        v8f acc0, acc1;
        #pragma unroll
        for (int e = 0; e < 8; ++e) { acc0[e] = b0; acc1[e] = b1; }

        const ushort_t* xrow = xbf + ((size_t)ln * kT + t) * kD;  // A row m = ln
        const ushort_t* hrow = hprev + (size_t)ln * kD;

        // pull next step's x row toward the caches while we compute this step
        if (t + 1 < kT) __builtin_prefetch(xrow + kD, 0, 3);

        #pragma unroll 4
        for (int kb = 0; kb < kKcat / 32; ++kb) {
            const int ks = kb * 32;
            union { v16bf v; float4 f4[2]; } a, p0, p1;
            // A fragment: K 0..511 from x_t, 512..1023 from h_{t-1} (uniform select)
            const ushort_t* ab = (ks < kD) ? (xrow + ks) : (hrow + (ks - kD));
            a.f4[0]  = *(const float4*)(ab + hi * 8);
            a.f4[1]  = *(const float4*)(ab + 16 + hi * 8);
            // B fragments from LDS (N-major -> contiguous 16B reads)
            p0.f4[0] = *(const float4*)(wrow0 + ks + hi * 8);
            p0.f4[1] = *(const float4*)(wrow0 + ks + 16 + hi * 8);
            p1.f4[0] = *(const float4*)(wrow1 + ks + hi * 8);
            p1.f4[1] = *(const float4*)(wrow1 + ks + 16 + hi * 8);
            acc0 = __builtin_amdgcn_wmma_f32_16x16x32_bf16(false, a.v, false, p0.v,
                                                           (short)0, acc0, false, false);
            acc1 = __builtin_amdgcn_wmma_f32_16x16x32_bf16(false, a.v, false, p1.v,
                                                           (short)0, acc1, false, false);
        }

        // D layout: element e -> M = e + hi*8, N = ln (+16 for second tile)
        #pragma unroll
        for (int e = 0; e < 8; ++e) {
            const int m = e + hi * 8;
            z_lds[((size_t)w * kB + m) * kCols + ln]      = acc0[e];
            z_lds[((size_t)w * kB + m) * kCols + 16 + ln] = acc1[e];
        }
        __syncthreads();

        // elementwise gates + state update: 512 outputs / 128 threads
        #pragma unroll
        for (int r = 0; r < (kB * kCols) / kThreads; ++r) {
            const int idx = tid + r * kThreads;
            const int m = idx >> 5, n = idx & (kCols - 1);
            const float zi = z_lds[(0 * kB + m) * kCols + n];
            const float zf = z_lds[(1 * kB + m) * kCols + n];
            const float zg = z_lds[(2 * kB + m) * kCols + n];
            const float zo = z_lds[(3 * kB + m) * kCols + n];
            const float ig = fast_sigmoid(zi);
            const float fg = fast_sigmoid(zf);
            const float gg = fast_tanh(zg);
            const float og = fast_sigmoid(zo);
            const float cn = __builtin_fmaf(fg, c_lds[idx], ig * gg);
            const float hn = og * fast_tanh(cn);
            c_lds[idx] = cn;
            hnext[(size_t)m * kD + d0 + n] = f2bf(hn);
            atomicAdd(out + ((size_t)m * kT + t) * kD + d0 + n, hn);  // sum over layers
            if (t == kT - 1) {
                out[HT_OFF + ((size_t)l * kB + m) * kD + d0 + n] = hn;
                out[CT_OFF + ((size_t)l * kB + m) * kD + d0 + n] = cn;
            }
        }

        // device-scope per-layer barrier: monotonic counter, release/acquire
        __threadfence();
        __syncthreads();
        if (tid == 0) {
            __hip_atomic_fetch_add(&bar[l], 1u, __ATOMIC_RELEASE, __HIP_MEMORY_SCOPE_AGENT);
            const unsigned tgt = (unsigned)(t + 1) * kNWG;
            while (__hip_atomic_load(&bar[l], __ATOMIC_ACQUIRE, __HIP_MEMORY_SCOPE_AGENT) < tgt)
                __builtin_amdgcn_s_sleep(2);
        }
        __syncthreads();
    }
}

extern "C" void kernel_launch(void* const* d_in, const int* in_sizes, int n_in,
                              void* d_out, int out_size, void* d_ws, size_t ws_size,
                              hipStream_t stream) {
    (void)in_sizes; (void)n_in; (void)out_size; (void)ws_size;
    const float* x    = (const float*)d_in[0];   // [B,T,D]
    const float* kern = (const float*)d_in[1];   // [L,D,4D]
    const float* rec  = (const float*)d_in[2];   // [L,D,4D]
    const float* bias = (const float*)d_in[3];   // [L,4D]
    float* out = (float*)d_out;
    char* ws = (char*)d_ws;
    ushort_t* xbf  = (ushort_t*)(ws + XBF_OFF);
    ushort_t* wcat = (ushort_t*)(ws + WCAT_OFF);
    ushort_t* hbuf = (ushort_t*)(ws + HBUF_OFF);
    unsigned* bar  = (unsigned*)(ws + BAR_OFF);

    prep_zero<<<1024, 256, 0, stream>>>(out, hbuf, bar);
    {
        const size_t n = (size_t)kB * kT * kD;
        cvt_x<<<(unsigned)((n + 255) / 256), 256, 0, stream>>>(x, xbf);
    }
    {
        const size_t n = (size_t)kL * k4D * kKcat;
        cvt_w<<<(unsigned)((n + 255) / 256), 256, 0, stream>>>(kern, rec, wcat);
    }
    lstm_scan<<<kL * kNWG, kThreads, SMEM_BYTES, stream>>>(xbf, wcat, bias, hbuf, bar, out);
}
